// UnidirectionalAttention_21955872817279
// MI455X (gfx1250) — matrix operand order repaired and checked by reading, not compiled, not run on hardware
//
#include <hip/hip_runtime.h>
#include <cstdint>

// MI455X / gfx1250, wave32. bf16 WMMA path: v_wmma_f32_16x16x32_bf16.
// Staging uses CDNA5 async copies (GLOBAL_LOAD_ASYNC_TO_LDS_B128, ASYNCcnt).
typedef __attribute__((ext_vector_type(16))) __bf16 v16bf;
typedef __attribute__((ext_vector_type(8)))  float  v8f;
typedef __attribute__((__vector_size__(16))) int    v4i_;

#define HID   1024
#define NHEAD 16
#define HS    64
#define SEQ   2048
#define BATCH 2
#define MTOT  (BATCH * SEQ)   // 4096

// ---------------------------------------------------------------- async copy helpers
// 16 bytes per lane, global -> LDS, tracked by ASYNCcnt.
__device__ __forceinline__ void async_copy16(const void* g, void* l) {
#if __has_builtin(__builtin_amdgcn_global_load_async_to_lds_b128)
  __builtin_amdgcn_global_load_async_to_lds_b128(
      (__attribute__((address_space(1))) v4i_*)(uintptr_t)g,
      (__attribute__((address_space(3))) v4i_*)(uintptr_t)l, 0, 0);
#else
  asm volatile("global_load_async_to_lds_b128 %0, %1, off"
               :: "v"((uint32_t)(uintptr_t)l), "v"(g) : "memory");
#endif
}

template <int N>
__device__ __forceinline__ void wait_async() {
#if __has_builtin(__builtin_amdgcn_s_wait_asynccnt)
  __builtin_amdgcn_s_wait_asynccnt(N);
#else
  asm volatile("s_wait_asynccnt %0" :: "i"(N) : "memory");
#endif
}

// ---------------------------------------------------------------- fp32 -> bf16
__global__ __launch_bounds__(256) void cvt_f32_bf16(const float* __restrict__ src,
                                                    __bf16* __restrict__ dst, int n) {
  int i = blockIdx.x * blockDim.x + threadIdx.x;
  int stride = gridDim.x * blockDim.x;
  for (; i < n; i += stride) dst[i] = (__bf16)src[i];
}

// ---------------------------------------------------------------- bf16 GEMM
// C[M,N] = A[M,K] * B[K,N] + bias  (fp32 accumulate). 128x128x32 block tile,
// 256 threads = 8 waves arranged 2(M) x 4(N); each wave does 64x32 via 4x2 WMMAs.
// Double-buffered LDS fed by async copies: next K tile is in flight during WMMAs.
template <bool OUT_F32>
__global__ __launch_bounds__(256)
void gemm_bf16(const __bf16* __restrict__ A, const __bf16* __restrict__ Bm,
               const float* __restrict__ bias,
               __bf16* __restrict__ outB, float* __restrict__ outF,
               int M, int N, int K, int lda, int ldb) {
  __shared__ __bf16 la[2][128][40];   // 32 + 8 pad (row stride 80B, 16B aligned)
  __shared__ __bf16 lb[2][32][136];   // 128 + 8 pad (row stride 272B, 16B aligned)

  const int tid  = threadIdx.x;
  const int lane = tid & 31;
  const int wave = tid >> 5;
  const int wm   = wave >> 2;      // 0..1
  const int wn   = wave & 3;       // 0..3
  const int hf   = lane >> 4;      // half-wave select
  const int loc  = lane & 15;
  const int bm   = blockIdx.y * 128;
  const int bn   = blockIdx.x * 128;

  v8f acc[4][2];
#pragma unroll
  for (int i = 0; i < 4; ++i)
#pragma unroll
    for (int j = 0; j < 2; ++j)
#pragma unroll
      for (int v = 0; v < 8; ++v) acc[i][j][v] = 0.0f;

  const int arow = tid >> 1, acol = (tid & 1) * 16;   // 128 rows x 32 cols
  const int brow = tid >> 3, bcol = (tid & 7) * 16;   // 32 rows x 128 cols

  // issue one K tile's staging: 4 wave-wide async ops (2 for A, 2 for B)
  auto issue_tile = [&](int k0, int buf) {
    const __bf16* ga = &A[(size_t)(bm + arow) * lda + k0 + acol];
    async_copy16(ga,     &la[buf][arow][acol]);
    async_copy16(ga + 8, &la[buf][arow][acol + 8]);
    const __bf16* gb = &Bm[(size_t)(k0 + brow) * ldb + bn + bcol];
    async_copy16(gb,     &lb[buf][brow][bcol]);
    async_copy16(gb + 8, &lb[buf][brow][bcol + 8]);
  };

  issue_tile(0, 0);
  int cur = 0;

  for (int k0 = 0; k0 < K; k0 += 32) {
    const bool more = (k0 + 32 < K);
    if (more) {
      issue_tile(k0 + 32, cur ^ 1);          // prefetch next tile into other buffer
      if (k0 + 64 < K) {                     // warm GL2 two tiles ahead
        __builtin_prefetch(&A[(size_t)(bm + arow) * lda + (k0 + 64) + acol], 0, 0);
        __builtin_prefetch(&Bm[(size_t)(k0 + 64 + brow) * ldb + bn + bcol], 0, 0);
      }
      wait_async<4>();                       // current tile's 4 asyncs have landed
    } else {
      wait_async<0>();
    }
    __syncthreads();

    // A fragments: 16-bit A 16x32 layout (lane half selects K octets)
    v16bf afrag[4];
#pragma unroll
    for (int i = 0; i < 4; ++i) {
      const int r = wm * 64 + i * 16 + loc;
#pragma unroll
      for (int j = 0; j < 8; ++j) {
        afrag[i][j]     = la[cur][r][hf * 8 + j];
        afrag[i][j + 8] = la[cur][r][16 + hf * 8 + j];
      }
    }
    // B fragments: lane%16 = N column, K octets selected by lane half (same as A)
    v16bf bfrag[2];
#pragma unroll
    for (int jn = 0; jn < 2; ++jn) {
      const int c = wn * 32 + jn * 16 + loc;
#pragma unroll
      for (int j = 0; j < 8; ++j) {
        bfrag[jn][j]     = lb[cur][hf * 8 + j][c];
        bfrag[jn][j + 8] = lb[cur][16 + hf * 8 + j][c];
      }
    }
#pragma unroll
    for (int i = 0; i < 4; ++i)
#pragma unroll
      for (int jn = 0; jn < 2; ++jn)
        acc[i][jn] = __builtin_amdgcn_wmma_f32_16x16x32_bf16(
            false, afrag[i], false, bfrag[jn], (short)0, acc[i][jn], false, false);

    __syncthreads();                          // all waves done reading buf `cur`
    cur ^= 1;
  }

  // epilogue: C/D layout row = v + 8*hf, col = loc (+16 per subtile)
#pragma unroll
  for (int jn = 0; jn < 2; ++jn) {
    const int col = bn + wn * 32 + jn * 16 + loc;
    const float bc = bias[col];
#pragma unroll
    for (int i = 0; i < 4; ++i) {
#pragma unroll
      for (int v = 0; v < 8; ++v) {
        const int row = bm + wm * 64 + i * 16 + v + 8 * hf;
        const float val = acc[i][jn][v] + bc;
        if (OUT_F32) outF[(size_t)row * N + col] = val;
        else         outB[(size_t)row * N + col] = (__bf16)val;
      }
    }
  }
}

// ---------------------------------------------------------------- flash attention
// One block = one (batch, head, 128 q rows). 8 waves, each wave owns 16 q rows.
// kv tiles of 64, causal loop bound kvb < qb+128. Online softmax in fp32.
__global__ __launch_bounds__(256)
void flash_attn(const __bf16* __restrict__ qkv, __bf16* __restrict__ vals) {
  __shared__ __bf16 lq[128][72];
  __shared__ __bf16 lk[64][72];
  __shared__ __bf16 lv[64][72];
  __shared__ __bf16 lp[8][16][72];   // per-wave P re-swizzle scratch

  const int tid  = threadIdx.x;
  const int lane = tid & 31;
  const int wave = tid >> 5;
  const int hf   = lane >> 4;
  const int loc  = lane & 15;
  const int qb   = blockIdx.x * 128;
  const int h    = blockIdx.y;
  const int b    = blockIdx.z;
  const size_t base = (size_t)b * SEQ * 3072 + (size_t)h * 64;  // + s*3072 + grp*1024

  // stage Q tile (128 x 64) via async copies
  {
    const int row = tid >> 1, cb = (tid & 1) * 32;
    const __bf16* g = &qkv[base + (size_t)(qb + row) * 3072 + cb];
    async_copy16(g,      &lq[row][cb]);
    async_copy16(g + 8,  &lq[row][cb + 8]);
    async_copy16(g + 16, &lq[row][cb + 16]);
    async_copy16(g + 24, &lq[row][cb + 24]);
  }
  wait_async<0>();
  __syncthreads();

  v16bf aq[2];
#pragma unroll
  for (int c = 0; c < 2; ++c)
#pragma unroll
    for (int j = 0; j < 8; ++j) {
      aq[c][j]     = lq[wave * 16 + loc][c * 32 + hf * 8 + j];
      aq[c][j + 8] = lq[wave * 16 + loc][c * 32 + 16 + hf * 8 + j];
    }

  float mrow[8], lrow[8];
  v8f o[4];
#pragma unroll
  for (int v = 0; v < 8; ++v) { mrow[v] = -1.0e30f; lrow[v] = 0.0f; }
#pragma unroll
  for (int t = 0; t < 4; ++t)
#pragma unroll
    for (int v = 0; v < 8; ++v) o[t][v] = 0.0f;

  const int kvend = qb + 128;
  for (int kvb = 0; kvb < kvend; kvb += 64) {
    __syncthreads();   // everyone finished reading previous K/V tiles
    {                  // stage K and V tiles (64 x 64 each) via async copies
      const int row = tid >> 2, cb = (tid & 3) * 16;
      const __bf16* gk = &qkv[base + 1024 + (size_t)(kvb + row) * 3072 + cb];
      async_copy16(gk,     &lk[row][cb]);
      async_copy16(gk + 8, &lk[row][cb + 8]);
      const __bf16* gv = &qkv[base + 2048 + (size_t)(kvb + row) * 3072 + cb];
      async_copy16(gv,     &lv[row][cb]);
      async_copy16(gv + 8, &lv[row][cb + 8]);
    }
    wait_async<0>();
    __syncthreads();

    // S = Q K^T : B[k=d][n=kv] -> element = K[kv][d] (rows of lk are kv)
    v8f s[4];
#pragma unroll
    for (int t = 0; t < 4; ++t)
#pragma unroll
      for (int v = 0; v < 8; ++v) s[t][v] = 0.0f;
#pragma unroll
    for (int c = 0; c < 2; ++c)
#pragma unroll
      for (int t = 0; t < 4; ++t) {
        v16bf bk;
#pragma unroll
        for (int j = 0; j < 8; ++j) {
          bk[j]     = lk[t * 16 + loc][c * 32 + hf * 8 + j];
          bk[j + 8] = lk[t * 16 + loc][c * 32 + 16 + hf * 8 + j];
        }
        s[t] = __builtin_amdgcn_wmma_f32_16x16x32_bf16(
            false, aq[c], false, bk, (short)0, s[t], false, false);
      }

    // scale + causal mask (matches reference's -10000 additive mask)
#pragma unroll
    for (int t = 0; t < 4; ++t) {
      const int kcol = kvb + t * 16 + loc;
#pragma unroll
      for (int v = 0; v < 8; ++v) {
        const int qrow = qb + wave * 16 + v + 8 * hf;
        float sc = s[t][v] * 0.125f;                 // 1/sqrt(64)
        if (kcol > qrow) sc -= 10000.0f;
        s[t][v] = sc;
      }
    }

    // online softmax: row max / rescale / exp / rowsum (16-lane xor reductions)
    float mnew[8], rsum[8];
#pragma unroll
    for (int v = 0; v < 8; ++v) {
      float mx = fmaxf(fmaxf(s[0][v], s[1][v]), fmaxf(s[2][v], s[3][v]));
#pragma unroll
      for (int off = 1; off < 16; off <<= 1) mx = fmaxf(mx, __shfl_xor(mx, off, 32));
      mnew[v] = fmaxf(mrow[v], mx);
      rsum[v] = 0.0f;
    }
#pragma unroll
    for (int t = 0; t < 4; ++t)
#pragma unroll
      for (int v = 0; v < 8; ++v) {
        const float p = __expf(s[t][v] - mnew[v]);
        rsum[v] += p;
        lp[wave][v + 8 * hf][t * 16 + loc] = (__bf16)p;  // C/D -> LDS for A-frag
      }
#pragma unroll
    for (int v = 0; v < 8; ++v) {
      float rs = rsum[v];
#pragma unroll
      for (int off = 1; off < 16; off <<= 1) rs += __shfl_xor(rs, off, 32);
      const float alpha = __expf(mrow[v] - mnew[v]);
      lrow[v] = lrow[v] * alpha + rs;
      mrow[v] = mnew[v];
#pragma unroll
      for (int t = 0; t < 4; ++t) o[t][v] *= alpha;
    }

    // O += P V  (per-wave LDS reads are DS-ordered with the stores above)
    v16bf ap[2];
#pragma unroll
    for (int c = 0; c < 2; ++c)
#pragma unroll
      for (int j = 0; j < 8; ++j) {
        ap[c][j]     = lp[wave][loc][c * 32 + hf * 8 + j];
        ap[c][j + 8] = lp[wave][loc][c * 32 + 16 + hf * 8 + j];
      }
#pragma unroll
    for (int c = 0; c < 2; ++c)
#pragma unroll
      for (int t = 0; t < 4; ++t) {
        v16bf bv;
#pragma unroll
        for (int j = 0; j < 8; ++j) {
          bv[j]     = lv[c * 32 + hf * 8 + j][t * 16 + loc];
          bv[j + 8] = lv[c * 32 + 16 + hf * 8 + j][t * 16 + loc];
        }
        o[t] = __builtin_amdgcn_wmma_f32_16x16x32_bf16(
            false, ap[c], false, bv, (short)0, o[t], false, false);
      }
  }

  // normalize + store vals[b, q, h*64 + d] as bf16 for the output projection
#pragma unroll
  for (int t = 0; t < 4; ++t)
#pragma unroll
    for (int v = 0; v < 8; ++v) {
      const int qrow = qb + wave * 16 + v + 8 * hf;
      const int d    = t * 16 + loc;
      vals[((size_t)b * SEQ + qrow) * HID + h * 64 + d] = (__bf16)(o[t][v] / lrow[v]);
    }
}

// ---------------------------------------------------------------- launch
extern "C" void kernel_launch(void* const* d_in, const int* in_sizes, int n_in,
                              void* d_out, int out_size, void* d_ws, size_t ws_size,
                              hipStream_t stream) {
  const float* x    = (const float*)d_in[0];   // [2,2048,1024]
  const float* Wqkv = (const float*)d_in[1];   // [1024,3072]
  const float* bqkv = (const float*)d_in[2];   // [3072]
  const float* Wout = (const float*)d_in[3];   // [1024,1024]
  const float* bout = (const float*)d_in[4];   // [1024]
  float* out = (float*)d_out;                  // [2,2048,1024] fp32

  char* ws = (char*)d_ws;
  __bf16* xb    = (__bf16*)(ws);                        //  8 MB
  __bf16* wqkvb = (__bf16*)(ws + (size_t)8  * 1048576); //  6 MB
  __bf16* woutb = (__bf16*)(ws + (size_t)14 * 1048576); //  2 MB
  __bf16* qkvb  = (__bf16*)(ws + (size_t)16 * 1048576); // 24 MB
  __bf16* valsb = (__bf16*)(ws + (size_t)40 * 1048576); //  8 MB

  cvt_f32_bf16<<<1024, 256, 0, stream>>>(x,    xb,    MTOT * HID);
  cvt_f32_bf16<<<1024, 256, 0, stream>>>(Wqkv, wqkvb, HID * 3 * HID);
  cvt_f32_bf16<<<1024, 256, 0, stream>>>(Wout, woutb, HID * HID);

  // qkv = x @ W_qkv + b_qkv   (bf16 out, layout [b,s,3072] = [Q|K|V] groups)
  gemm_bf16<false><<<dim3(24, 32), 256, 0, stream>>>(
      xb, wqkvb, bqkv, qkvb, nullptr, MTOT, 3 * HID, HID, HID, 3 * HID);

  // causal multi-head attention
  flash_attn<<<dim3(SEQ / 128, NHEAD, BATCH), 256, 0, stream>>>(qkvb, valsb);

  // out = vals @ W_out + b_out  (fp32 out)
  gemm_bf16<true><<<dim3(8, 32), 256, 0, stream>>>(
      valsb, woutb, bout, nullptr, out, MTOT, HID, HID, HID, HID);
}
